// PolarAttention_68101001445725
// MI455X (gfx1250) — compile-verified, matrix operand order checked by reading, not run on hardware
//
#include <hip/hip_runtime.h>

typedef __attribute__((ext_vector_type(16))) _Float16 v16h;
typedef __attribute__((ext_vector_type(8)))  _Float16 v8h;
typedef __attribute__((ext_vector_type(8)))  float    v8f;

#define N_HEADS 4
#define DH      32
#define D       128
#define INNER   128
#define BN      12
#define TQ      1024
#define TK      2112
#define LN_EPS  1e-5f
#define NEG_BIG (-3.402823466e38f)

static __device__ __forceinline__ v16h cat16(v8h lo, v8h hi) {
  return __builtin_shufflevector(lo, hi, 0, 1, 2, 3, 4, 5, 6, 7,
                                 8, 9, 10, 11, 12, 13, 14, 15);
}

// Load a contiguous-row A/B fragment: halves [kb..kb+7] and [16+kb..16+kb+7].
static __device__ __forceinline__ v16h frag_row(const _Float16* rowp, int kb) {
  v8h lo = *(const v8h*)(rowp + kb);
  v8h hi = *(const v8h*)(rowp + 16 + kb);
  return cat16(lo, hi);
}

// ---------------------------------------------------------------------------
// Kernel 1: convert the four fp32 weights (128x128) to f16 in FRAGMENT-READY
// swizzled order: idx = ((kc*8 + ct)*32 + lane)*16 + j, where
//   n   = ct*16 + (lane&15), kb = (lane>>4)*8,
//   row = kc*32 + (j<8 ? kb + j : 16 + kb + (j-8)).
// A B-fragment load is then one contiguous 32-byte v16h read per lane.
// ---------------------------------------------------------------------------
__global__ __launch_bounds__(256)
void pa_cvt_weights(const float* __restrict__ Wq, const float* __restrict__ Wk,
                    const float* __restrict__ Wv, const float* __restrict__ Wo,
                    _Float16* __restrict__ dst) {
  const int i = blockIdx.x * 256 + threadIdx.x;     // 0 .. 65535
  const int mat = i >> 14;
  const int r   = i & 16383;
  const int j    = r & 15;
  const int lane = (r >> 4) & 31;
  const int ct   = (r >> 9) & 7;
  const int kc   = r >> 12;
  const int n    = ct * 16 + (lane & 15);
  const int kb   = (lane >> 4) * 8;
  const int row  = kc * 32 + ((j < 8) ? (kb + j) : (16 + kb + (j - 8)));
  const float* W = (mat == 0) ? Wq : (mat == 1) ? Wk : (mat == 2) ? Wv : Wo;
  dst[i] = (_Float16)W[row * 128 + n];
}

// ---------------------------------------------------------------------------
// Kernel 2: LayerNorm (d=128 per token) + 128x128 projection, WMMA f16->f32.
// x: (BN, D, T).  out: (BN, T, INNER) f16, or (BN, INNER, T) if transposeOut.
// `scale` folds the attention dh^-0.5 into the Q projection.
// ---------------------------------------------------------------------------
__global__ __launch_bounds__(256)
void pa_ln_proj(const float* __restrict__ x,
                const float* __restrict__ gma, const float* __restrict__ bta,
                const _Float16* __restrict__ Wz,    // swizzled (16384 halves)
                const float* __restrict__ bias,
                _Float16* __restrict__ out,
                int T, float scale, int transposeOut) {
  __shared__ __align__(16) _Float16 xs[64][D];      // 16 KB normalized tokens

  const int tid = threadIdx.x;
  const int bn  = blockIdx.y;
  const int t0  = blockIdx.x * 64;

  if (tid < 64) {
    const int t = t0 + tid;
    const float* xp = x + (size_t)bn * D * T + t;
    float s = 0.f, ss = 0.f;
    #pragma unroll 4
    for (int c = 0; c < D; ++c) { float v = xp[(size_t)c * T]; s += v; ss += v * v; }
    const float mu  = s * (1.f / D);
    const float var = ss * (1.f / D) - mu * mu;
    const float rs  = rsqrtf(var + LN_EPS);
    #pragma unroll 4
    for (int c = 0; c < D; ++c) {
      float v = xp[(size_t)c * T];
      xs[tid][c] = (_Float16)(((v - mu) * rs) * gma[c] + bta[c]);
    }
  }
  __syncthreads();

  const int lane  = tid & 31;
  const int wv    = tid >> 5;
  const int rt    = wv >> 1;        // row tile (16 tokens)
  const int chalf = wv & 1;         // column half (64 cols)
  const int mrow  = lane & 15;
  const int kb    = (lane >> 4) * 8;
  const int hi16  = lane >> 4;

  v8f zero = {};
  v8f acc[4];
  #pragma unroll
  for (int ct = 0; ct < 4; ++ct) acc[ct] = zero;

  #pragma unroll
  for (int kc = 0; kc < 4; ++kc) {
    const v16h a = frag_row(&xs[rt * 16 + mrow][kc * 32], kb);
    #pragma unroll
    for (int ct = 0; ct < 4; ++ct) {
      const int ctg = chalf * 4 + ct;
      const v16h b = *(const v16h*)(Wz + ((((kc * 8) + ctg) * 32 + lane) << 4));
      acc[ct] = __builtin_amdgcn_wmma_f32_16x16x32_f16(false, a, false, b,
                                                       (short)0, acc[ct], false, false);
    }
  }

  if (transposeOut) {
    // out[(bn, col, token)] with 8 consecutive tokens per lane -> v8h store.
    #pragma unroll
    for (int ct = 0; ct < 4; ++ct) {
      const int col = chalf * 64 + ct * 16 + mrow;
      const float bc = bias[col];
      v8h ov;
      #pragma unroll
      for (int r = 0; r < 8; ++r) ov[r] = (_Float16)((acc[ct][r] + bc) * scale);
      *(v8h*)(out + ((size_t)bn * INNER + col) * T + t0 + rt * 16 + 8 * hi16) = ov;
    }
  } else {
    #pragma unroll
    for (int ct = 0; ct < 4; ++ct) {
      const int col = chalf * 64 + ct * 16 + mrow;
      const float bc = bias[col];
      #pragma unroll
      for (int r = 0; r < 8; ++r) {
        const int m = rt * 16 + r + 8 * hi16;
        out[((size_t)bn * T + t0 + m) * INNER + col] =
            (_Float16)((acc[ct][r] + bc) * scale);
      }
    }
  }
}

// ---------------------------------------------------------------------------
// Kernel 3: flash attention, TRANSPOSED score trick.
//   S^T = K x Q^T  : A = K chunk (16k x 32dh, rows contiguous in kp),
//                    B = Q^T (32dh x 16q, rows contiguous in qp), resident.
//   C-fragment of S^T: lane holds column q = lane&15, rows k in VGPR index
//   -> softmax is in-lane + one shfl_xor(16) to merge lane-halves, and the
//   P^T B-fragment for PV is EXACTLY the exp'd C-fragment (no LDS transpose).
//   O^T = V^T x P^T : A = V^T (dh-major V, rows contiguous), two M-tiles.
// ---------------------------------------------------------------------------
__global__ __launch_bounds__(256)
void pa_attention(const _Float16* __restrict__ qp,  // (BN, TQ, INNER), pre-scaled
                  const _Float16* __restrict__ kp,  // (BN, TK, INNER)
                  const _Float16* __restrict__ vt,  // (BN, INNER, TK)  dh-major
                  _Float16* __restrict__ ao) {      // (BN, TQ, INNER)
  const int tid  = threadIdx.x;
  const int lane = tid & 31;
  const int wv   = tid >> 5;
  const int h    = blockIdx.y;
  const int bn   = blockIdx.z;
  const int q0   = (blockIdx.x * 8 + wv) * 16;

  const int nlow = lane & 15;
  const int kb   = (lane >> 4) * 8;
  const int hi16 = lane >> 4;

  // Resident Q^T B-fragment (32dh x 16q): lane column q = nlow.
  const v16h bq =
      frag_row(qp + ((size_t)bn * TQ + q0 + nlow) * INNER + h * DH, kb);

  v8f zero = {};
  v8f o0 = zero, o1 = zero;
  float rmax = NEG_BIG, rsum = 0.f;

  const _Float16* kbase = kp + (size_t)bn * TK * INNER + h * DH;
  const _Float16* vrow0 = vt + ((size_t)bn * INNER + h * DH + nlow) * TK;
  const _Float16* vrow1 = vrow0 + (size_t)16 * TK;

  for (int t0 = 0; t0 < TK; t0 += 32) {
    // ---- S^T: K A-fragments, row m = k-token ----
    const _Float16* krow0 = kbase + (size_t)(t0 + nlow) * INNER;
    if (t0 + 32 < TK) __builtin_prefetch(krow0 + (size_t)32 * INNER, 0, 1);
    const v16h ak0 = frag_row(krow0, kb);
    const v16h ak1 = frag_row(krow0 + (size_t)16 * INNER, kb);
    v8f s0 = __builtin_amdgcn_wmma_f32_16x16x32_f16(false, ak0, false, bq,
                                                    (short)0, zero, false, false);
    v8f s1 = __builtin_amdgcn_wmma_f32_16x16x32_f16(false, ak1, false, bq,
                                                    (short)0, zero, false, false);

    // ---- online softmax (per-lane over k, one shfl to merge halves) ----
    float mx = fmaxf(s0[0], s1[0]);
    #pragma unroll
    for (int j = 1; j < 8; ++j) mx = fmaxf(mx, fmaxf(s0[j], s1[j]));
    mx = fmaxf(mx, __shfl_xor(mx, 16, 32));
    const float mnew  = fmaxf(rmax, mx);
    const float alpha = __expf(rmax - mnew);
    rmax = mnew;

    v16h bp;                        // P^T B-fragment, directly from registers
    float ls = 0.f;
    #pragma unroll
    for (int j = 0; j < 8; ++j) {
      const float e0 = __expf(s0[j] - mnew);
      const float e1 = __expf(s1[j] - mnew);
      ls += e0 + e1;
      bp[j]     = (_Float16)e0;
      bp[8 + j] = (_Float16)e1;
    }
    ls += __shfl_xor(ls, 16, 32);
    rsum = rsum * alpha + ls;
    #pragma unroll
    for (int j = 0; j < 8; ++j) { o0[j] *= alpha; o1[j] *= alpha; }

    // ---- O^T += V^T x P^T : V^T A-fragments (dh rows contiguous) ----
    const v16h av0 = frag_row(vrow0 + t0, kb);
    const v16h av1 = frag_row(vrow1 + t0, kb);
    o0 = __builtin_amdgcn_wmma_f32_16x16x32_f16(false, av0, false, bp,
                                                (short)0, o0, false, false);
    o1 = __builtin_amdgcn_wmma_f32_16x16x32_f16(false, av1, false, bp,
                                                (short)0, o1, false, false);
  }

  // O^T fragment: lane column q = nlow, dh = r + 8*hi16 (+16 for o1).
  const float inv = 1.f / rsum;
  _Float16* orow = ao + ((size_t)bn * TQ + q0 + nlow) * INNER + h * DH;
  v8h w0, w1;
  #pragma unroll
  for (int r = 0; r < 8; ++r) {
    w0[r] = (_Float16)(o0[r] * inv);
    w1[r] = (_Float16)(o1[r] * inv);
  }
  *(v8h*)(orow + 8 * hi16)      = w0;
  *(v8h*)(orow + 16 + 8 * hi16) = w1;
}

// ---------------------------------------------------------------------------
// Kernel 4: z = a @ Wo + bo + skip, stored transposed as (BN, D, TQ) fp32.
// ---------------------------------------------------------------------------
__global__ __launch_bounds__(256)
void pa_out_proj(const _Float16* __restrict__ ai,   // (BN, TQ, INNER)
                 const _Float16* __restrict__ Wz,   // swizzled Wo
                 const float* __restrict__ bo,
                 const float* __restrict__ skip,    // (BN, D, TQ)
                 float* __restrict__ out) {         // (BN, D, TQ)
  const int tid = threadIdx.x;
  const int bn  = blockIdx.y;
  const int t0  = blockIdx.x * 64;

  const int lane  = tid & 31;
  const int wv    = tid >> 5;
  const int rt    = wv >> 1;
  const int chalf = wv & 1;
  const int mrow  = lane & 15;
  const int kb    = (lane >> 4) * 8;
  const int hi16  = lane >> 4;

  v8f zero = {};
  v8f acc[4];
  #pragma unroll
  for (int ct = 0; ct < 4; ++ct) acc[ct] = zero;

  #pragma unroll
  for (int kc = 0; kc < 4; ++kc) {
    const v16h a = frag_row(
        ai + ((size_t)bn * TQ + t0 + rt * 16 + mrow) * INNER + kc * 32, kb);
    #pragma unroll
    for (int ct = 0; ct < 4; ++ct) {
      const int ctg = chalf * 4 + ct;
      const v16h b = *(const v16h*)(Wz + ((((kc * 8) + ctg) * 32 + lane) << 4));
      acc[ct] = __builtin_amdgcn_wmma_f32_16x16x32_f16(false, a, false, b,
                                                       (short)0, acc[ct], false, false);
    }
  }

  #pragma unroll
  for (int ct = 0; ct < 4; ++ct) {
    const int col = chalf * 64 + ct * 16 + mrow;
    const float bc = bo[col];
    #pragma unroll
    for (int r = 0; r < 8; ++r) {
      const int t = t0 + rt * 16 + r + 8 * hi16;
      const size_t idx = ((size_t)bn * D + col) * TQ + t;
      out[idx] = acc[ct][r] + bc + skip[idx];
    }
  }
}

// ---------------------------------------------------------------------------
extern "C" void kernel_launch(void* const* d_in, const int* in_sizes, int n_in,
                              void* d_out, int out_size, void* d_ws, size_t ws_size,
                              hipStream_t stream) {
  const float* q     = (const float*)d_in[0];
  const float* k     = (const float*)d_in[1];
  const float* v     = (const float*)d_in[2];
  const float* skip  = (const float*)d_in[3];
  const float* lnqg  = (const float*)d_in[4];
  const float* lnqb  = (const float*)d_in[5];
  const float* lnkg  = (const float*)d_in[6];
  const float* lnkb  = (const float*)d_in[7];
  const float* lnvg  = (const float*)d_in[8];
  const float* lnvb  = (const float*)d_in[9];
  const float* Wq    = (const float*)d_in[10];
  const float* bq    = (const float*)d_in[11];
  const float* Wk    = (const float*)d_in[12];
  const float* bk    = (const float*)d_in[13];
  const float* Wv    = (const float*)d_in[14];
  const float* bv    = (const float*)d_in[15];
  const float* Wo    = (const float*)d_in[16];
  const float* bo    = (const float*)d_in[17];
  float* out = (float*)d_out;

  char* ws = (char*)d_ws;
  _Float16* W16 = (_Float16*)ws;                       // 4 x 16384 halves, swizzled
  _Float16* qp  = (_Float16*)(ws + 131072);            // (BN, TQ, INNER)
  _Float16* kp  = qp + (size_t)BN * TQ * INNER;        // (BN, TK, INNER)
  _Float16* vt  = kp + (size_t)BN * TK * INNER;        // (BN, INNER, TK) dh-major
  _Float16* ao  = vt + (size_t)BN * TK * INNER;        // (BN, TQ, INNER)

  const float qscale = 0.17677669529663689f;           // dh^-0.5

  pa_cvt_weights<<<dim3(256), dim3(256), 0, stream>>>(Wq, Wk, Wv, Wo, W16);

  pa_ln_proj<<<dim3(TQ / 64, BN), dim3(256), 0, stream>>>(
      q, lnqg, lnqb, W16, bq, qp, TQ, qscale, 0);
  pa_ln_proj<<<dim3(TK / 64, BN), dim3(256), 0, stream>>>(
      k, lnkg, lnkb, W16 + 16384, bk, kp, TK, 1.0f, 0);
  pa_ln_proj<<<dim3(TK / 64, BN), dim3(256), 0, stream>>>(
      v, lnvg, lnvb, W16 + 32768, bv, vt, TK, 1.0f, 1);

  pa_attention<<<dim3(TQ / 128, N_HEADS, BN), dim3(256), 0, stream>>>(
      qp, kp, vt, ao);

  pa_out_proj<<<dim3(TQ / 64, BN), dim3(256), 0, stream>>>(
      ao, W16 + 49152, bo, skip, out);
}
